// OTModel_9981503996091
// MI455X (gfx1250) — matrix operand order repaired and checked by reading, not compiled, not run on hardware
//
#include <hip/hip_runtime.h>

#define NN 4096
#define BB 512

typedef _Float16 h16;
typedef __attribute__((ext_vector_type(16))) _Float16 v16h;
typedef __attribute__((ext_vector_type(8)))  _Float16 v8h;
typedef __attribute__((ext_vector_type(8)))  float    v8f;
typedef int i32x4 __attribute__((vector_size(16)));

// ---------------- optional CDNA5 async global->LDS copy ----------------
#if defined(__has_builtin)
#if __has_builtin(__builtin_amdgcn_global_load_async_to_lds_b128) && \
    __has_builtin(__builtin_amdgcn_s_wait_asynccnt)
#define USE_ASYNC_COPY 1
#endif
#endif
#ifndef USE_ASYNC_COPY
#define USE_ASYNC_COPY 0
#endif

#if USE_ASYNC_COPY
// param types per hipcc diagnostic: (int4 AS1*, int4 AS3*, imm int, imm int)
#define AS1V(p) ((__attribute__((address_space(1))) i32x4*)(void*)(p))
#define AS3V(p) ((__attribute__((address_space(3))) i32x4*)(void*)(p))
#endif

// ---------------- workspace layout (bytes) ----------------
static constexpr size_t SZ_K16  = (size_t)NN * NN * 2;          // 32 MB
static constexpr size_t SZ_NB4  = (size_t)NN * BB * 4;          // 8 MB
static constexpr size_t SZ_NB2  = (size_t)NN * BB * 2;          // 4 MB
static constexpr size_t OFF_K16   = 0;
static constexpr size_t OFF_KT16  = OFF_K16  + SZ_K16;
static constexpr size_t OFF_XT    = OFF_KT16 + SZ_K16;          // [3*BB, NN] f16
static constexpr size_t OFF_KF    = OFF_XT   + (size_t)3 * BB * NN * 2;
static constexpr size_t OFF_KFV   = OFF_KF   + SZ_NB4;          // Kf
static constexpr size_t OFF_KFFL  = OFF_KFV  + SZ_NB4;          // K@(f*flog)
static constexpr size_t OFF_A16   = OFF_KFFL + SZ_NB4;
static constexpr size_t OFF_R16   = OFF_A16  + SZ_NB2;
static constexpr size_t OFF_YT    = OFF_R16  + SZ_NB2;          // [2*BB, NN] f16
static constexpr size_t OFF_KTA   = OFF_YT   + (size_t)2 * BB * NN * 2;
static constexpr size_t OFF_KTR   = OFF_KTA  + SZ_NB4;
static constexpr size_t OFF_SCALE = OFF_KTR  + SZ_NB4;          // [512] f32
static constexpr size_t OFF_COLP  = OFF_SCALE + 4096;           // [512][512] f32
static constexpr size_t OFF_TERMP = OFF_COLP + (size_t)512 * 512 * 4; // [512][3]
static constexpr size_t OFF_KLP   = OFF_TERMP + 512 * 3 * 4;    // [512]

// ---------------- helpers ----------------
__device__ __forceinline__ float block_reduce_sum(float v, float* red) {
  const int tid = threadIdx.x;
  red[tid] = v;
  __syncthreads();
  for (int off = 128; off > 0; off >>= 1) {
    if (tid < off) red[tid] += red[tid + off];
    __syncthreads();
  }
  float r = red[0];
  __syncthreads();
  return r;
}

// ---------------- K -> f16 + f16 transpose ----------------
__global__ __launch_bounds__(256) void k_convK(const float* __restrict__ K,
                                               h16* __restrict__ K16,
                                               h16* __restrict__ KT16) {
  __shared__ h16 t[32][33];
  const int tx = threadIdx.x, ty = threadIdx.y;
  const int x0 = blockIdx.x * 32, y0 = blockIdx.y * 32;
#pragma unroll
  for (int r = 0; r < 4; ++r) {
    const int y = y0 + ty + 8 * r;
    const h16 h = (h16)K[(size_t)y * NN + x0 + tx];
    K16[(size_t)y * NN + x0 + tx] = h;
    t[ty + 8 * r][tx] = h;
  }
  __syncthreads();
#pragma unroll
  for (int r = 0; r < 4; ++r) {
    const int y = x0 + ty + 8 * r;
    KT16[(size_t)y * NN + y0 + tx] = t[tx][ty + 8 * r];
  }
}

// ---------------- column partial sums of flog ----------------
__global__ __launch_bounds__(256) void k_colsum(const float* __restrict__ U,
                                                float* __restrict__ colpart,
                                                const int* epsP, const int* lamP) {
  const float lam = (float)*lamP, eps = (float)*epsP;
  const float loe = lam / eps;
  const int b = blockIdx.x, tid = threadIdx.x;
  float s0 = 0.f, s1 = 0.f;
#pragma unroll
  for (int r = 0; r < 8; ++r) {
    const size_t base = (size_t)(8 * b + r) * BB;
    s0 += loe * logf(lam / (lam - U[base + tid]));
    s1 += loe * logf(lam / (lam - U[base + tid + 256]));
  }
  colpart[(size_t)tid * 512 + b]         = s0;
  colpart[(size_t)(tid + 256) * 512 + b] = s1;
}

__global__ __launch_bounds__(256) void k_scale(const float* __restrict__ colpart,
                                               float* __restrict__ scale) {
  __shared__ float red[256];
  const int j = blockIdx.x, tid = threadIdx.x;
  float s = colpart[(size_t)j * 512 + tid] + colpart[(size_t)j * 512 + tid + 256];
  s = block_reduce_sum(s, red);
  if (tid == 0) scale[j] = s * (1.0f / NN);
}

// ---------------- build XT = [e^T | f^T | (f*flog)^T] in f16 ----------------
__global__ __launch_bounds__(256) void k_buildXT(const float* __restrict__ U,
                                                 const float* __restrict__ scale,
                                                 h16* __restrict__ XT,
                                                 const int* epsP, const int* lamP) {
  __shared__ h16 te[32][33], tf[32][33], tg[32][33];
  const float lam = (float)*lamP, eps = (float)*epsP;
  const float loe = lam / eps;
  const int tx = threadIdx.x, ty = threadIdx.y;
  const int i0 = blockIdx.x * 32, j0 = blockIdx.y * 32;
#pragma unroll
  for (int r = 0; r < 4; ++r) {
    const int i = i0 + ty + 8 * r, j = j0 + tx;
    const float u    = U[(size_t)i * BB + j];
    const float flog = loe * logf(lam / (lam - u));
    const float f    = expf(flog);
    te[ty + 8 * r][tx] = (h16)expf(flog - scale[j]);
    tf[ty + 8 * r][tx] = (h16)f;
    tg[ty + 8 * r][tx] = (h16)(f * flog);
  }
  __syncthreads();
#pragma unroll
  for (int r = 0; r < 4; ++r) {
    const int rowX = j0 + ty + 8 * r, col = i0 + tx;
    XT[(size_t)rowX * NN + col]            = te[tx][ty + 8 * r];
    XT[(size_t)(BB + rowX) * NN + col]     = tf[tx][ty + 8 * r];
    XT[(size_t)(2 * BB + rowX) * NN + col] = tg[tx][ty + 8 * r];
  }
}

// ---------------- WMMA f16 GEMM, LDS-staged + double buffered ---------------
// out = A (row-major [NN,NN] f16) @ RHS, where BT holds RHS transposed
// ([cols, NN] row-major f16).  Block tile 64x256, 8 waves, wave tile 32x64.
// Output column c -> o[c>>9] local column c&511, row-major [NN,BB] f32.
#define BM 64
#define BN 256
#define BK 32
#define KSTEPS (NN / BK)

__global__ __launch_bounds__(256) void k_gemm_f16(const h16* __restrict__ A,
                                                  const h16* __restrict__ BT,
                                                  float* __restrict__ o0,
                                                  float* __restrict__ o1,
                                                  float* __restrict__ o2) {
  __shared__ __align__(16) h16 As[2][BM][40];   // 80B row stride (bank de-skew)
  __shared__ __align__(16) h16 Bs[2][BN][40];

  const int tid  = threadIdx.x;
  const int lane = tid & 31;
  const int wave = tid >> 5;
  const int wm = wave & 1;        // 2 wave rows
  const int wn = wave >> 1;       // 4 wave cols
  const int m  = lane & 15;
  const int hi = lane >> 4;       // lane group (ISA 16-bit fragment layouts)
  const int c0 = hi * 8;          // A fragment k sub-offset
  const int kb = hi * 16;         // unused; B fragment uses c0 like A

  const int rowBase = blockIdx.x * BM;
  const int colBase = blockIdx.y * BN;

  // cooperative fill mapping: A = 64 rows x 32 halves, B = 256 rows x 32 halves
  const int ar = tid >> 2;              // 0..63
  const int ac = (tid & 3) * 8;         // halves
  const h16* gA = A  + (size_t)(rowBase + ar)  * NN + ac;
  const h16* gB = BT + (size_t)(colBase + tid) * NN;

#if !USE_ASYNC_COPY
  v8h  rA;
  v16h rB0, rB1;
#endif

  auto issue_fill = [&](int s, int k0) {
#if USE_ASYNC_COPY
    __builtin_amdgcn_global_load_async_to_lds_b128(AS1V(gA + k0),      AS3V(&As[s][ar][ac]),  0, 0);
    __builtin_amdgcn_global_load_async_to_lds_b128(AS1V(gB + k0),      AS3V(&Bs[s][tid][0]),  0, 0);
    __builtin_amdgcn_global_load_async_to_lds_b128(AS1V(gB + k0 + 8),  AS3V(&Bs[s][tid][8]),  0, 0);
    __builtin_amdgcn_global_load_async_to_lds_b128(AS1V(gB + k0 + 16), AS3V(&Bs[s][tid][16]), 0, 0);
    __builtin_amdgcn_global_load_async_to_lds_b128(AS1V(gB + k0 + 24), AS3V(&Bs[s][tid][24]), 0, 0);
    __builtin_prefetch(gB + k0 + 64, 0, 1);   // gfx1250 global_prefetch_b8
#else
    (void)s;
    rA  = *(const v8h*)(gA + k0);
    rB0 = *(const v16h*)(gB + k0);
    rB1 = *(const v16h*)(gB + k0 + 16);
    __builtin_prefetch(gB + k0 + 64, 0, 1);   // gfx1250 global_prefetch_b8
#endif
  };

  auto finish_fill = [&](int s) {
#if USE_ASYNC_COPY
    (void)s;
    __builtin_amdgcn_s_wait_asynccnt(0);
#else
    *(v8h*)&As[s][ar][ac]  = rA;
    *(v8h*)&Bs[s][tid][0]  = *((v8h*)&rB0);
    *(v8h*)&Bs[s][tid][8]  = *(((v8h*)&rB0) + 1);
    *(v8h*)&Bs[s][tid][16] = *((v8h*)&rB1);
    *(v8h*)&Bs[s][tid][24] = *(((v8h*)&rB1) + 1);
#endif
  };

  v8f acc[2][4] = {};

  // prologue: stage 0
  issue_fill(0, 0);
  finish_fill(0);
  __syncthreads();

  for (int kt = 0; kt < KSTEPS; ++kt) {
    const int s = kt & 1;
    const bool more = (kt + 1) < KSTEPS;

    // start next tile's global->LDS transfers early (overlap with 8 WMMAs)
    if (more) issue_fill(s ^ 1, (kt + 1) * BK);

    // A fragments (2 row tiles) from LDS
    v16h af[2];
#pragma unroll
    for (int mt = 0; mt < 2; ++mt) {
      const int arow = wm * 32 + mt * 16 + m;
      *(v8h*)&af[mt]         = *(const v8h*)&As[s][arow][c0];
      *(((v8h*)&af[mt]) + 1) = *(const v8h*)&As[s][arow][16 + c0];
    }
#pragma unroll
    for (int t = 0; t < 4; ++t) {
      const int brow = wn * 64 + t * 16 + m;
      v16h bf;
      *(v8h*)&bf         = *(const v8h*)&Bs[s][brow][c0];
      *(((v8h*)&bf) + 1) = *(const v8h*)&Bs[s][brow][16 + c0];
      acc[0][t] = __builtin_amdgcn_wmma_f32_16x16x32_f16(false, af[0], false, bf, (short)0, acc[0][t], false, false);
      acc[1][t] = __builtin_amdgcn_wmma_f32_16x16x32_f16(false, af[1], false, bf, (short)0, acc[1][t], false, false);
    }

    if (more) finish_fill(s ^ 1);
    __syncthreads();
  }

  const int colg = colBase + wn * 64;
  const int arr  = colg >> 9;
  const int lc   = (colg & 511) + m;
  float* o = (arr == 0) ? o0 : ((arr == 1) ? o1 : o2);
#pragma unroll
  for (int mt = 0; mt < 2; ++mt) {
    const int rb = rowBase + wm * 32 + mt * 16 + hi * 8;  // C/D: lanes 16-31 hold M=8..15
#pragma unroll
    for (int g = 0; g < 8; ++g) {
      const size_t rowoff = (size_t)(rb + g) * BB;
      o[rowoff + lc +  0] = acc[mt][0][g];
      o[rowoff + lc + 16] = acc[mt][1][g];
      o[rowoff + lc + 32] = acc[mt][2][g];
      o[rowoff + lc + 48] = acc[mt][3][g];
    }
  }
}

// ---------------- elementwise after GEMM1: a, r=P/Kf, term1..3 partials -----
__global__ __launch_bounds__(256) void k_post1(const float* __restrict__ P,
                                               const float* __restrict__ KF,
                                               const float* __restrict__ Kf,
                                               const float* __restrict__ Kffl,
                                               const float* __restrict__ scale,
                                               h16* __restrict__ a16,
                                               h16* __restrict__ r16,
                                               float* __restrict__ termpart) {
  __shared__ float red[768];
  const int b = blockIdx.x, tid = threadIdx.x;
  float t1 = 0.f, t2 = 0.f, t3 = 0.f;
#pragma unroll 4
  for (int r = 0; r < 16; ++r) {
    const size_t t = (size_t)b * 4096 + tid + 256 * r;
    const int j = (int)(t & 511);
    const float p   = P[t];
    const float kfs = KF[t];
    const float kf  = Kf[t];
    const float kff = Kffl[t];
    const float aoe = logf(p) - logf(kfs) - scale[j];
    const float a   = expf(aoe);
    t1 += a * kff;
    t2 += a * aoe * kf;
    t3 += a * kf;
    a16[t] = (h16)a;
    r16[t] = (h16)(p / kf);
  }
  red[tid] = t1; red[256 + tid] = t2; red[512 + tid] = t3;
  __syncthreads();
  for (int off = 128; off > 0; off >>= 1) {
    if (tid < off) {
      red[tid]       += red[tid + off];
      red[256 + tid] += red[256 + tid + off];
      red[512 + tid] += red[512 + tid + off];
    }
    __syncthreads();
  }
  if (tid == 0) {
    termpart[3 * b + 0] = red[0];
    termpart[3 * b + 1] = red[256];
    termpart[3 * b + 2] = red[512];
  }
}

// ---------------- f16 transpose [NN,BB] -> rows of YT [2*BB, NN] ------------
__global__ __launch_bounds__(256) void k_transY(const h16* __restrict__ a16,
                                                const h16* __restrict__ r16,
                                                h16* __restrict__ YT) {
  __shared__ h16 t[32][33];
  const int tx = threadIdx.x, ty = threadIdx.y;
  const int i0 = blockIdx.x * 32, j0 = blockIdx.y * 32;
  const int z = blockIdx.z;
  const h16* in = z ? r16 : a16;
#pragma unroll
  for (int r = 0; r < 4; ++r)
    t[ty + 8 * r][tx] = in[(size_t)(i0 + ty + 8 * r) * BB + j0 + tx];
  __syncthreads();
#pragma unroll
  for (int r = 0; r < 4; ++r)
    YT[(size_t)(z * BB + j0 + ty + 8 * r) * NN + i0 + tx] = t[tx][ty + 8 * r];
}

// ---------------- elementwise after GEMM2: KL partials ----------------------
__global__ __launch_bounds__(256) void k_post2(const float* __restrict__ U,
                                               const float* __restrict__ KTa,
                                               const float* __restrict__ KTr,
                                               float* __restrict__ klpart,
                                               const int* epsP, const int* lamP) {
  __shared__ float red[256];
  const float lam = (float)*lamP, eps = (float)*epsP;
  const float loe = lam / eps;
  const int b = blockIdx.x, tid = threadIdx.x;
  float kl = 0.f;
#pragma unroll 4
  for (int r = 0; r < 16; ++r) {
    const size_t t = (size_t)b * 4096 + tid + 256 * r;
    const float u = U[t];
    const float L = logf(lam / (lam - u));
    const float gcol = expf(loe * L) * KTa[t];          // f * (K^T a)
    const float q    = expf((loe + 1.0f) * L) * KTr[t]; // pow * (K^T (P/Kf))
    kl += gcol * logf(gcol / q) - gcol;
  }
  kl = block_reduce_sum(kl, red);
  if (tid == 0) klpart[b] = kl;
}

// ---------------- final deterministic reduction -----------------------------
__global__ __launch_bounds__(256) void k_final(const float* __restrict__ termpart,
                                               const float* __restrict__ klpart,
                                               const int* epsP, const int* lamP,
                                               float* __restrict__ out) {
  __shared__ float red[256];
  const float lam = (float)*lamP, eps = (float)*epsP;
  const int tid = threadIdx.x;
  float t1 = 0.f, t2 = 0.f, t3 = 0.f, kl = 0.f;
  for (int b = tid; b < 512; b += 256) {
    t1 += termpart[3 * b + 0];
    t2 += termpart[3 * b + 1];
    t3 += termpart[3 * b + 2];
    kl += klpart[b];
  }
  t1 = block_reduce_sum(t1, red);
  t2 = block_reduce_sum(t2, red);
  t3 = block_reduce_sum(t3, red);
  kl = block_reduce_sum(kl, red);
  if (tid == 0) out[0] = eps * (t1 + t2 - t3) + lam * kl;
}

// ---------------- host launch ----------------
extern "C" void kernel_launch(void* const* d_in, const int* in_sizes, int n_in,
                              void* d_out, int out_size, void* d_ws, size_t ws_size,
                              hipStream_t stream) {
  (void)in_sizes; (void)n_in; (void)out_size; (void)ws_size;
  const float* U = (const float*)d_in[0];
  const float* P = (const float*)d_in[1];
  const float* K = (const float*)d_in[2];
  const int* epsP = (const int*)d_in[3];
  const int* lamP = (const int*)d_in[4];
  char* w = (char*)d_ws;

  h16*   K16   = (h16*)  (w + OFF_K16);
  h16*   KT16  = (h16*)  (w + OFF_KT16);
  h16*   XT    = (h16*)  (w + OFF_XT);
  float* KF    = (float*)(w + OFF_KF);
  float* Kfv   = (float*)(w + OFF_KFV);
  float* Kffl  = (float*)(w + OFF_KFFL);
  h16*   a16   = (h16*)  (w + OFF_A16);
  h16*   r16   = (h16*)  (w + OFF_R16);
  h16*   YT    = (h16*)  (w + OFF_YT);
  float* KTa   = (float*)(w + OFF_KTA);
  float* KTr   = (float*)(w + OFF_KTR);
  float* scale = (float*)(w + OFF_SCALE);
  float* colp  = (float*)(w + OFF_COLP);
  float* termp = (float*)(w + OFF_TERMP);
  float* klp   = (float*)(w + OFF_KLP);

  const dim3 b32(32, 8);

  k_convK  <<<dim3(128, 128), b32, 0, stream>>>(K, K16, KT16);
  k_colsum <<<512, 256, 0, stream>>>(U, colp, epsP, lamP);
  k_scale  <<<512, 256, 0, stream>>>(colp, scale);
  k_buildXT<<<dim3(128, 16), b32, 0, stream>>>(U, scale, XT, epsP, lamP);
  // GEMM1: [KF | Kf | Kffl] = K16 @ [e | f | f*flog]   (1536 columns)
  k_gemm_f16<<<dim3(64, 6), 256, 0, stream>>>(K16, XT, KF, Kfv, Kffl);
  k_post1  <<<512, 256, 0, stream>>>(P, KF, Kfv, Kffl, scale, a16, r16, termp);
  k_transY <<<dim3(128, 16, 2), b32, 0, stream>>>(a16, r16, YT);
  // GEMM2: [KTa | KTr] = K^T16 @ [a | P/Kf]            (1024 columns)
  k_gemm_f16<<<dim3(64, 4), 256, 0, stream>>>(KT16, YT, KTa, KTr, KTr);
  k_post2  <<<512, 256, 0, stream>>>(U, KTa, KTr, klp, epsP, lamP);
  k_final  <<<1, 256, 0, stream>>>(termp, klp, epsP, lamP, (float*)d_out);
}
